// GATLayer_84146999263289
// MI455X (gfx1250) — matrix operand order repaired and checked by reading, not compiled, not run on hardware
//
#include <hip/hip_runtime.h>

#define B_ 8
#define N_ 1024
#define INF_ 256
#define NH_ 4
#define DF_ 64
#define NEG_SLOPE_ 0.2f

#define JCHUNK 256                      // j columns staged per phase
#define NPHASE (N_ / JCHUNK)            // 4
#define WROW 264                        // padded LDS row stride (elems): bank stride 4
#define LWHT_ELEMS (DF_ * WROW)         // 16896 elems per buffer
#define LDS_BYTES (2 * LWHT_ELEMS * 2 + N_ * 4)   // 2 WhT buffers + full s_dst row = 71680B

typedef __attribute__((ext_vector_type(16))) __bf16 v16bf;
typedef __attribute__((ext_vector_type(8)))  float  v8f;

#if defined(__HIP_DEVICE_COMPILE__) && __has_builtin(__builtin_amdgcn_tensor_load_to_lds)
#define HAVE_TDM 1
#if defined(__clang_major__) && (__clang_major__ >= 23)
#define TDM_SIX_ARGS 1
#endif
#else
#define HAVE_TDM 0
#endif

#if HAVE_TDM
typedef __attribute__((ext_vector_type(4))) unsigned int tdm_u32x4;
typedef __attribute__((ext_vector_type(8))) int          tdm_i32x8;
typedef __attribute__((ext_vector_type(4))) int          tdm_i32x4;

// issue a 2D TDM load: tile 256(j) x 64(d) of 2B elems, tensor row stride 1024,
// LDS padding 16B per 512B row -> LDS row stride = WROW elems (conflict-free ds reads)
__device__ __forceinline__ void tdm_stage(unsigned int ldsa, unsigned long long ga) {
    tdm_u32x4 g0;
    g0[0] = 1u;                                                  // count=1 (valid user D#)
    g0[1] = ldsa;                                                // lds_addr
    g0[2] = (unsigned int)(ga & 0xFFFFFFFFull);                  // global_addr lo
    g0[3] = (unsigned int)((ga >> 32) & 0x1FFFFFFu) | (2u << 30);// addr hi | type=2 (image)
    tdm_i32x8 g1;
    g1[0] = (int)((1u << 16)      // data_size = 2B
                | (1u << 20)      // pad_enable
                | (6u << 22)      // pad_interval: 128 dwords (one 256-elem row)
                | (3u << 25));    // pad_amount: 4 dwords (16B)
    g1[1] = (int)(((unsigned)N_) << 16);       // tensor_dim0 lo16 = 1024
    g1[2] = (int)(((unsigned)DF_) << 16);      // tensor_dim0 hi | tensor_dim1 lo16 = 64
    g1[3] = (int)(((unsigned)JCHUNK) << 16);   // tensor_dim1 hi | tile_dim0 = 256
    g1[4] = (int)DF_;                          // tile_dim1 = 64, tile_dim2 = 0
    g1[5] = (int)N_;                           // tensor_dim0_stride lo32 = 1024
    g1[6] = 0;
    g1[7] = 0;
    tdm_i32x4 z4 = {0, 0, 0, 0};
#ifdef TDM_SIX_ARGS
    tdm_i32x8 z8 = {0, 0, 0, 0, 0, 0, 0, 0};
    __builtin_amdgcn_tensor_load_to_lds(g0, g1, z4, z4, z8, 0);
#else
    __builtin_amdgcn_tensor_load_to_lds(g0, g1, z4, z4, 0);
#endif
}
#endif

union AB16 {
    v16bf v;
    uint4 q[2];
    unsigned short s[16];
};

__device__ __forceinline__ unsigned short f2bf(float f) {
    unsigned int u = __float_as_uint(f);
    u += 0x7FFFu + ((u >> 16) & 1u);   // round-to-nearest-even
    return (unsigned short)(u >> 16);
}
__device__ __forceinline__ float bf2f(unsigned short s) {
    return __uint_as_float(((unsigned int)s) << 16);
}

// ---------------- conversion kernels ----------------
__global__ void cvt_h_kernel(const float* __restrict__ h, unsigned short* __restrict__ hb, int n) {
    int i = blockIdx.x * blockDim.x + threadIdx.x;
    if (i < n) hb[i] = f2bf(h[i]);
}

__global__ void cvt_wt_kernel(const float* __restrict__ W, unsigned short* __restrict__ WT) {
    int i = blockIdx.x * blockDim.x + threadIdx.x;   // 65536
    int c = i >> 8, k = i & 255;
    WT[i] = f2bf(W[k * 256 + c]);
}

__global__ void adj_mask_kernel(const int* __restrict__ adj, unsigned int* __restrict__ maskadj) {
    int t = blockIdx.x * blockDim.x + threadIdx.x;   // 32768
    int i = t >> 5, w = t & 31;
    const int* row = adj + (size_t)i * N_ + w * 32;
    unsigned int bits = 0;
    #pragma unroll
    for (int q = 0; q < 32; q += 4) {
        int4 v = *(const int4*)(row + q);
        bits |= (v.x != 0 ? 1u : 0u) << (q + 0);
        bits |= (v.y != 0 ? 1u : 0u) << (q + 1);
        bits |= (v.z != 0 ? 1u : 0u) << (q + 2);
        bits |= (v.w != 0 ? 1u : 0u) << (q + 3);
    }
    maskadj[t] = bits;
}

// ---------------- phase 1: Wh = h @ W  -> WhT[b][h][d][n] (bf16) ----------------
__global__ __launch_bounds__(256) void wh_gemm_kernel(
    const unsigned short* __restrict__ hb,
    const unsigned short* __restrict__ WT,
    unsigned short* __restrict__ WhT)
{
    int tid  = blockIdx.x * blockDim.x + threadIdx.x;
    int w    = tid >> 5;
    int lane = tid & 31;
    int mt = w >> 4;
    int nt = w & 15;
    int r  = lane & 15;
    int g  = lane >> 4;

    const unsigned short* arow = hb + (size_t)(mt * 16 + r) * INF_;
    int c = nt * 16 + r;
    const unsigned short* brow = WT + (size_t)c * INF_;

    v8f acc = {};
    #pragma unroll
    for (int kk = 0; kk < 8; ++kk) {
        int kb = kk * 32;
        AB16 A, Bm;
        A.q[0]  = *(const uint4*)(arow + kb + g * 8);
        A.q[1]  = *(const uint4*)(arow + kb + g * 8 + 16);
        Bm.q[0] = *(const uint4*)(brow + kb + g * 8);
        Bm.q[1] = *(const uint4*)(brow + kb + g * 8 + 16);
        acc = __builtin_amdgcn_wmma_f32_16x16x32_bf16(false, A.v, false, Bm.v,
                                                      (short)0, acc, false, false);
    }

    int b  = mt >> 6;
    int hh = c >> 6;
    int d  = c & 63;
    int n0 = (mt & 63) * 16 + g * 8;
    unsigned short tmp[8];
    #pragma unroll
    for (int v = 0; v < 8; ++v) tmp[v] = f2bf(acc[v]);
    *(uint4*)(WhT + ((size_t)((b * NH_ + hh) * DF_ + d)) * N_ + n0) = *(const uint4*)tmp;
}

// ---------------- s_src / s_dst = Wh . a ----------------
__global__ void s_kernel(const unsigned short* __restrict__ WhT,
                         const float* __restrict__ a,
                         float* __restrict__ s_src, float* __restrict__ s_dst)
{
    int i = blockIdx.x * blockDim.x + threadIdx.x;   // 32768
    int n  = i & (N_ - 1);
    int bh = i >> 10;
    int hh = bh & (NH_ - 1);
    const unsigned short* base = WhT + (size_t)bh * DF_ * N_ + n;
    float ss = 0.f, sd = 0.f;
    #pragma unroll 8
    for (int d = 0; d < DF_; ++d) {
        float v = bf2f(base[(size_t)d * N_]);
        ss += v * a[hh * 128 + d];
        sd += v * a[hh * 128 + 64 + d];
    }
    s_src[i] = ss;
    s_dst[i] = sd;
}

// ---------------- phase 2: fused masked-softmax attention (flash), TDM-staged LDS ----------------
__global__ __launch_bounds__(256) void gat_attn_kernel(
    const unsigned int* __restrict__ maskadj,
    const unsigned short* __restrict__ WhT,
    const float* __restrict__ s_src,
    const float* __restrict__ s_dst,
    float* __restrict__ out)
{
    extern __shared__ char ldsbuf[];
    unsigned short* lds_wht = (unsigned short*)ldsbuf;                 // 2 x [DF][WROW]
    float* lds_sd = (float*)(ldsbuf + 2 * LWHT_ELEMS * 2);             // [N_] full s_dst row

    int tid  = threadIdx.x;
    int wave = tid >> 5;
    int lane = tid & 31;
    int wg = blockIdx.x;
    int bh = wg >> 3;
    int it = (wg & 7) * 8 + wave;
    int b  = bh >> 2;
    int hh = bh & 3;
    int r = lane & 15;
    int g = lane >> 4;

    float ssrc = s_src[bh * N_ + it * 16 + r];
    const unsigned int* maskrow = maskadj + (size_t)(it * 16 + r) * 32;
    const unsigned short* whtb = WhT + (size_t)bh * DF_ * N_;
    const float* sdrow = s_dst + bh * N_;

    // stage the full s_dst row once (4KB, 256 threads x float4)
    *(float4*)(lds_sd + tid * 4) = *(const float4*)(sdrow + tid * 4);

#if HAVE_TDM
    if (tid < 32)
        tdm_stage((unsigned int)(size_t)lds_wht, (unsigned long long)(size_t)whtb);
#endif

    float m = -1e30f, l = 0.f;
    v8f acc0 = {}, acc1 = {}, acc2 = {}, acc3 = {};

    for (int ph = 0; ph < NPHASE; ++ph) {
        int cur = ph & 1;
        unsigned short* buf = lds_wht + cur * LWHT_ELEMS;
#if HAVE_TDM
        if (tid < 32) __builtin_amdgcn_s_wait_tensorcnt(0);
        __syncthreads();                         // chunk ph ready; prev reads of other buffer done
        if ((ph + 1 < NPHASE) && (tid < 32))     // async-prefetch next chunk, overlaps compute
            tdm_stage((unsigned int)(size_t)(lds_wht + (1 - cur) * LWHT_ELEMS),
                      (unsigned long long)(size_t)whtb + (unsigned long long)(ph + 1) * JCHUNK * 2ull);
#else
        __syncthreads();                         // previous phase consumption done
        {
            int d = tid >> 2, part = tid & 3;    // 64 rows x 4 parts of 64 elems
            const unsigned short* src = whtb + (size_t)d * N_ + ph * JCHUNK + part * 64;
            unsigned short* dst = buf + d * WROW + part * 64;
            #pragma unroll
            for (int k2 = 0; k2 < 8; ++k2)
                *(uint4*)(dst + k2 * 8) = *(const uint4*)(src + k2 * 8);
        }
        __syncthreads();
#endif

        #pragma unroll
        for (int jt = 0; jt < JCHUNK / 32; ++jt) {
            unsigned int mw = maskrow[ph * (JCHUNK / 32) + jt];
            int jbg = ph * JCHUNK + jt * 32 + g * 8;     // global j base for this lane
            float4 sd0a = *(const float4*)(lds_sd + jbg);
            float4 sd0b = *(const float4*)(lds_sd + jbg + 4);
            float4 sd1a = *(const float4*)(lds_sd + jbg + 16);
            float4 sd1b = *(const float4*)(lds_sd + jbg + 20);
            float sdv[16] = {sd0a.x, sd0a.y, sd0a.z, sd0a.w, sd0b.x, sd0b.y, sd0b.z, sd0b.w,
                             sd1a.x, sd1a.y, sd1a.z, sd1a.w, sd1b.x, sd1b.y, sd1b.z, sd1b.w};
            unsigned int m1 = mw >> (g * 8);
            unsigned int m2 = mw >> (g * 8 + 16);

            float e[16];
            float rowmax = -1e30f;
            #pragma unroll
            for (int q = 0; q < 16; ++q) {
                unsigned int bit = (q < 8) ? ((m1 >> q) & 1u) : ((m2 >> (q - 8)) & 1u);
                float t = ssrc + sdv[q];
                t = (t > 0.f) ? t : NEG_SLOPE_ * t;
                t = bit ? t : -1e30f;
                e[q] = t;
                rowmax = fmaxf(rowmax, t);
            }
            rowmax = fmaxf(rowmax, __shfl_xor(rowmax, 16, 32));
            float mn  = fmaxf(m, rowmax);
            float fac = __expf(m - mn);
            m = mn;

            float psum = 0.f;
            AB16 P;
            #pragma unroll
            for (int q = 0; q < 16; ++q) {
                float p = (e[q] <= -1e29f) ? 0.f : __expf(e[q] - mn);
                psum += p;
                P.v[q] = (__bf16)p;
            }
            psum += __shfl_xor(psum, 16, 32);
            l = l * fac + psum;

            #pragma unroll
            for (int v = 0; v < 8; ++v) {
                float fv = __shfl(fac, v + g * 8, 32);
                acc0[v] *= fv; acc1[v] *= fv; acc2[v] *= fv; acc3[v] *= fv;
            }

            const unsigned short* lb = buf + jt * 32 + g * 8;
            AB16 B0, B1, B2, B3;
            B0.q[0] = *(const uint4*)(lb + (0  + r) * WROW);
            B0.q[1] = *(const uint4*)(lb + (0  + r) * WROW + 16);
            B1.q[0] = *(const uint4*)(lb + (16 + r) * WROW);
            B1.q[1] = *(const uint4*)(lb + (16 + r) * WROW + 16);
            B2.q[0] = *(const uint4*)(lb + (32 + r) * WROW);
            B2.q[1] = *(const uint4*)(lb + (32 + r) * WROW + 16);
            B3.q[0] = *(const uint4*)(lb + (48 + r) * WROW);
            B3.q[1] = *(const uint4*)(lb + (48 + r) * WROW + 16);

            acc0 = __builtin_amdgcn_wmma_f32_16x16x32_bf16(false, P.v, false, B0.v, (short)0, acc0, false, false);
            acc1 = __builtin_amdgcn_wmma_f32_16x16x32_bf16(false, P.v, false, B1.v, (short)0, acc1, false, false);
            acc2 = __builtin_amdgcn_wmma_f32_16x16x32_bf16(false, P.v, false, B2.v, (short)0, acc2, false, false);
            acc3 = __builtin_amdgcn_wmma_f32_16x16x32_bf16(false, P.v, false, B3.v, (short)0, acc3, false, false);
        }
#if !HAVE_TDM
        // manual path overwrites the buffer next phase; barrier at loop top protects
#endif
    }

    #pragma unroll
    for (int v = 0; v < 8; ++v) {
        float lv   = __shfl(l, v + g * 8, 32);
        float rinv = (lv > 0.f) ? (1.0f / lv) : 0.f;
        int n = it * 16 + v + g * 8;
        size_t ob = ((size_t)(b * N_ + n)) * (NH_ * DF_) + hh * DF_;
        float x0 = acc0[v] * rinv;
        float x1 = acc1[v] * rinv;
        float x2 = acc2[v] * rinv;
        float x3 = acc3[v] * rinv;
        x0 = (x0 > 0.f) ? x0 : (__expf(x0) - 1.f);
        x1 = (x1 > 0.f) ? x1 : (__expf(x1) - 1.f);
        x2 = (x2 > 0.f) ? x2 : (__expf(x2) - 1.f);
        x3 = (x3 > 0.f) ? x3 : (__expf(x3) - 1.f);
        out[ob + 0  + r] = x0;
        out[ob + 16 + r] = x1;
        out[ob + 32 + r] = x2;
        out[ob + 48 + r] = x3;
    }
}

extern "C" void kernel_launch(void* const* d_in, const int* in_sizes, int n_in,
                              void* d_out, int out_size, void* d_ws, size_t ws_size,
                              hipStream_t stream) {
    const float* h   = (const float*)d_in[0];
    const int*   adj = (const int*)d_in[1];
    const float* W   = (const float*)d_in[2];
    const float* a   = (const float*)d_in[3];
    float* out = (float*)d_out;

    char* ws = (char*)d_ws;
    unsigned short* hb    = (unsigned short*)ws;                               // 4MB
    unsigned short* WT    = (unsigned short*)(ws + (4u << 20));                // 128KB
    unsigned short* WhT   = (unsigned short*)(ws + (4u << 20) + (128u << 10)); // 4MB
    float* s_src          = (float*)(ws + (8u << 20) + (128u << 10));          // 128KB
    float* s_dst          = s_src + B_ * NH_ * N_;                             // 128KB
    unsigned int* maskadj = (unsigned int*)(ws + (8u << 20) + (384u << 10));   // 128KB

    const int nh_elems = B_ * N_ * INF_;
    cvt_h_kernel<<<(nh_elems + 255) / 256, 256, 0, stream>>>(h, hb, nh_elems);
    cvt_wt_kernel<<<(INF_ * INF_) / 256, 256, 0, stream>>>(W, WT);
    adj_mask_kernel<<<(N_ * 32) / 256, 256, 0, stream>>>(adj, maskadj);
    wh_gemm_kernel<<<1024, 256, 0, stream>>>(hb, WT, WhT);
    s_kernel<<<(B_ * NH_ * N_) / 256, 256, 0, stream>>>(WhT, a, s_src, s_dst);
    gat_attn_kernel<<<256, 256, LDS_BYTES, stream>>>(maskadj, WhT, s_src, s_dst, out);
}
